// MoELayer_72962904424643
// MI455X (gfx1250) — compile-verified, matrix operand order checked by reading, not run on hardware
//
#include <hip/hip_runtime.h>
#include <hip/hip_bf16.h>
#include <math.h>

#define NTOK   8192
#define CDIM   1024
#define DDIM   3072
#define NEXP   8
#define ROWS   (2 * NTOK)      // total compact (token,expert) rows = 2N always
#define TM     128
#define TN     128

// GEMM1: K=CDIM, N=DDIM ; GEMM2: K=DDIM, N=CDIM  (k-chunks of 32, n-tiles of 16)
#define KT1    (CDIM / 32)     // 32
#define NT1    (DDIM / 16)     // 192
#define KT2    (DDIM / 32)     // 96
#define NT2    (CDIM / 16)     // 64

typedef __bf16 v16bf __attribute__((ext_vector_type(16)));
typedef float  v8f   __attribute__((ext_vector_type(8)));

union FragBF { v16bf v; uint4 q[2]; };

#ifndef __has_builtin
#define __has_builtin(x) 0
#endif
#if __has_builtin(__builtin_amdgcn_tensor_load_to_lds) && __has_builtin(__builtin_amdgcn_s_wait_tensorcnt)
#define HAVE_TDM 1
#else
#define HAVE_TDM 0
#endif

__device__ __forceinline__ unsigned short f2bf(float f) {
    union { float f; unsigned int u; } x; x.f = f;
    unsigned int r = x.u + 0x7fffu + ((x.u >> 16) & 1u);   // round-to-nearest-even
    return (unsigned short)(r >> 16);
}

// Fragment-order tile: 1024B = [half][lane][16B].
// Lane l of a 16x32 bf16 fragment holds row/col = l&15, k-half = (l>>4)*8;
// q0 = k 0..7 (+khalf*8), q1 = k 16..23 (+khalf*8)  -> half 0 / half 1 regions.
__device__ __forceinline__ v16bf load_frag(const unsigned short* tileBase, int lane) {
    const unsigned short* p = tileBase + (lane << 3);
    FragBF f;
    f.q[0] = *(const uint4*)(p);          // half 0
    f.q[1] = *(const uint4*)(p + 256);    // half 1 (+512B)
    return f.v;
}

#if HAVE_TDM
typedef unsigned int u32x4 __attribute__((ext_vector_type(4)));
typedef int          i32x4 __attribute__((ext_vector_type(4)));
typedef int          i32x8 __attribute__((ext_vector_type(8)));

// 1-D TDM copy: 8KB contiguous global -> LDS (data_size = 8B, tile_dim0 = 1024 units)
__device__ __forceinline__ void tdm_load_8k(void* ldsDst, const void* gsrc) {
    unsigned long long ga = (unsigned long long)(size_t)gsrc;
    u32x4 g0;
    g0[0] = 1u;                                                  // count=1 (valid user D#)
    g0[1] = (unsigned int)(size_t)ldsDst;                        // lds_addr (bytes)
    g0[2] = (unsigned int)ga;                                    // global_addr[31:0]
    g0[3] = (unsigned int)((ga >> 32) & 0x01ffffffull)           // global_addr[56:32]
            | 0x80000000u;                                       // type = 2 ("image")
    i32x8 g1;
    g1[0] = (3 << 16);               // wg_mask=0, data_size=3 (8 bytes)
    g1[1] = 0;                       // tensor_dim0[15:0] (of 2^30) in [63:48] -> 0
    g1[2] = 0x4000 | (1 << 16);      // tensor_dim0[31:16]=0x4000 (2^30), tensor_dim1 lo = 1
    g1[3] = (1024 << 16);            // tensor_dim1 hi = 0, tile_dim0 = 1024 (8KB)
    g1[4] = 0;                       // tile_dim1 = 0 (unused), tile_dim2 = 0
    g1[5] = 1024;                    // tensor_dim0_stride lo (unused for 1 line)
    g1[6] = 0;
    g1[7] = 0;
    i32x4 z4 = {0, 0, 0, 0};
#if defined(__clang_major__) && (__clang_major__ >= 23)
    i32x8 z8 = {0, 0, 0, 0, 0, 0, 0, 0};
    __builtin_amdgcn_tensor_load_to_lds(g0, g1, z4, z4, z8, 0);
#else
    __builtin_amdgcn_tensor_load_to_lds(g0, g1, z4, z4, 0);
#endif
}
#endif

// ---------------------------------------------------------------- zero out
__global__ void zero_kernel(float* __restrict__ out, size_t n, int* __restrict__ counts) {
    size_t i = (size_t)blockIdx.x * blockDim.x + threadIdx.x;
    size_t stride = (size_t)gridDim.x * blockDim.x;
    for (; i < n; i += stride) out[i] = 0.0f;
    if (blockIdx.x == 0 && threadIdx.x < NEXP) counts[threadIdx.x] = 0;
}

// --------------------------- fp32 -> bf16 convert + weight fragment shuffle
// x: plain row-major bf16. W1/W2: per-(expert, k-chunk, n-tile) 1KB tiles in
// WMMA fragment order so GEMM staging is a pure contiguous copy.
__device__ __forceinline__ void shuffle_w_item(const float* __restrict__ W,
                                               unsigned short* __restrict__ Ws,
                                               int KT, int NT, int Ndim, size_t item) {
    int lane = (int)(item & 31);
    int half = (int)((item >> 5) & 1);
    size_t tile = item >> 6;                 // ((e*KT + tk)*NT + tn)
    int tn = (int)(tile % NT);
    size_t rest = tile / NT;
    int tk = (int)(rest % KT);
    int e  = (int)(rest / KT);
    int n = tn * 16 + (lane & 15);
    int kbase = tk * 32 + half * 16 + (lane >> 4) * 8;
    const float* src = W + ((size_t)e * KT * 32 + kbase) * Ndim + n;
    unsigned short v[8];
#pragma unroll
    for (int j = 0; j < 8; ++j) v[j] = f2bf(src[(size_t)j * Ndim]);
    *(uint4*)(Ws + tile * 512 + half * 256 + lane * 8) = *(const uint4*)v;
}

__global__ void convert_kernel(const float* __restrict__ x,
                               const float* __restrict__ W1,
                               const float* __restrict__ W2,
                               unsigned short* __restrict__ xbf,
                               unsigned short* __restrict__ w1s,
                               unsigned short* __restrict__ w2s) {
    const size_t IX  = (size_t)NTOK * CDIM / 8;              // 16B chunks of x
    const size_t IW1 = (size_t)NEXP * KT1 * NT1 * 64;        // (tile, half, lane) items
    const size_t IW2 = (size_t)NEXP * KT2 * NT2 * 64;
    const size_t total = IX + IW1 + IW2;
    size_t i = (size_t)blockIdx.x * blockDim.x + threadIdx.x;
    size_t stride = (size_t)gridDim.x * blockDim.x;
    for (; i < total; i += stride) {
        if (i < IX) {
            const float* s = x + i * 8;
            unsigned short v[8];
#pragma unroll
            for (int j = 0; j < 8; ++j) v[j] = f2bf(s[j]);
            *(uint4*)(xbf + i * 8) = *(const uint4*)v;
        } else if (i < IX + IW1) {
            shuffle_w_item(W1, w1s, KT1, NT1, DDIM, i - IX);
        } else {
            shuffle_w_item(W2, w2s, KT2, NT2, CDIM, i - IX - IW1);
        }
    }
}

// ------------------------------------------------------------------ router
__global__ void router_kernel(const float* __restrict__ x,
                              const float* __restrict__ rw,
                              int* __restrict__ t2e, float* __restrict__ t2w,
                              int* __restrict__ counts) {
    int wave = threadIdx.x >> 5, lane = threadIdx.x & 31;
    int t = blockIdx.x * 8 + wave;
    if (t >= NTOK) return;
    float acc[NEXP];
#pragma unroll
    for (int e = 0; e < NEXP; ++e) acc[e] = 0.0f;
    const float* xp = x + (size_t)t * CDIM;
    for (int c = lane; c < CDIM; c += 32) {
        float xv = xp[c];
#pragma unroll
        for (int e = 0; e < NEXP; ++e) acc[e] += xv * rw[e * CDIM + c];
    }
#pragma unroll
    for (int e = 0; e < NEXP; ++e) {
#pragma unroll
        for (int off = 16; off > 0; off >>= 1) acc[e] += __shfl_xor(acc[e], off, 32);
    }
    if (lane == 0) {
        float best = -3.4e38f, sec = -3.4e38f; int bi = 0, si = 0;
#pragma unroll
        for (int e = 0; e < NEXP; ++e) {
            float v = acc[e];
            if (v > best)     { sec = best; si = bi; best = v; bi = e; }
            else if (v > sec) { sec = v; si = e; }
        }
        float w0 = 1.0f / (1.0f + __expf(sec - best));
        t2e[2 * t] = bi;  t2e[2 * t + 1] = si;
        t2w[2 * t] = w0;  t2w[2 * t + 1] = 1.0f - w0;
        atomicAdd(&counts[bi], 1);
        atomicAdd(&counts[si], 1);
    }
}

// -------------------------------------------------------------------- scan
__global__ void scan_kernel(const int* __restrict__ counts,
                            int* __restrict__ offs, int* __restrict__ fill) {
    if (threadIdx.x == 0) {
        int s = 0;
        for (int e = 0; e < NEXP; ++e) { offs[e] = s; s += counts[e]; }
    }
    if (threadIdx.x < NEXP) fill[threadIdx.x] = 0;
}

// ------------------------------------------------------------------ assign
__global__ void assign_kernel(const int* __restrict__ t2e, const float* __restrict__ t2w,
                              const int* __restrict__ offs, int* __restrict__ fill,
                              int* __restrict__ rowtok, float* __restrict__ rowwt) {
    int t = blockIdx.x * blockDim.x + threadIdx.x;
    if (t >= NTOK) return;
#pragma unroll
    for (int k = 0; k < 2; ++k) {
        int e = t2e[2 * t + k];
        int slot = atomicAdd(&fill[e], 1);
        int row = offs[e] + slot;
        rowtok[row] = t;
        rowwt[row] = t2w[2 * t + k];
    }
}

// ------------------------------------------------- GEMM1: h = gelu(X W1+b1)
__global__ __launch_bounds__(256)
void gemm1_kernel(const unsigned short* __restrict__ xbf,
                  const unsigned short* __restrict__ w1s,
                  const float* __restrict__ b1,
                  unsigned short* __restrict__ hbuf,
                  const int* __restrict__ rowtok,
                  const int* __restrict__ counts,
                  const int* __restrict__ offs) {
    const int e = blockIdx.z;
    const int cnt = counts[e];
    const int mbase = blockIdx.x * TM;
    if (mbase >= cnt) return;                 // uniform exit: EXEC stays all-ones
    const int nbase = blockIdx.y * TN;
    const int off_e = offs[e];

    __shared__ unsigned short ldsA[8 * 512];       // 8 m-tiles, fragment order (8KB)
    __shared__ unsigned short ldsB[2][8 * 512];    // double-buffered B panel (2x8KB)
    __shared__ int ridx[TM];

    const int tid = threadIdx.x;
    const int lane = tid & 31, wave = tid >> 5;
    const int wm = wave & 3, wn = wave >> 2;       // 4x2 wave grid over 128x128 tile

    if (tid < TM) {
        int gm = mbase + tid;
        ridx[tid] = rowtok[off_e + (gm < cnt ? gm : 0)];
    }

    const v8f vzero = {0.f, 0.f, 0.f, 0.f, 0.f, 0.f, 0.f, 0.f};
    v8f acc[2][4];
#pragma unroll
    for (int i = 0; i < 2; ++i)
#pragma unroll
        for (int j = 0; j < 4; ++j) acc[i][j] = vzero;

    const size_t btile = (size_t)e * KT1 * NT1 + (nbase >> 4);   // B tile index @ kt=0
#if HAVE_TDM
    if (wave == 0) tdm_load_8k(&ldsB[0][0], w1s + btile * 512);
#endif

    for (int kt = 0; kt < KT1; ++kt) {
        __syncthreads();                           // previous compute done
        {   // stage A (gathered rows) into fragment order: 2x 16B per thread
            int r = tid >> 2;
            int kc = (tid & 3) << 3;               // 0,8,16,24
            int half = kc >> 4, khalf = (kc >> 3) & 1;
#pragma unroll
            for (int rep = 0; rep < 2; ++rep) {
                int rr = r + rep * 64;
                uint4 v = *(const uint4*)(xbf + (size_t)ridx[rr] * CDIM + kt * 32 + kc);
                int mt = rr >> 4, m = rr & 15;
                *(uint4*)(&ldsA[mt * 512 + half * 256 + (khalf * 16 + m) * 8]) = v;
            }
        }
#if HAVE_TDM
        if (wave == 0) {
            if (kt + 1 < KT1) {
                tdm_load_8k(&ldsB[(kt + 1) & 1][0], w1s + (btile + (size_t)(kt + 1) * NT1) * 512);
                __builtin_amdgcn_s_wait_tensorcnt(1);
            } else {
                __builtin_amdgcn_s_wait_tensorcnt(0);
            }
        }
#else
        {   // fallback: vectorized b128 copy of the 8KB fragment-order panel
            const uint4* gs = (const uint4*)(w1s + (btile + (size_t)kt * NT1) * 512);
            uint4* ds = (uint4*)(&ldsB[kt & 1][0]);
            ds[tid] = gs[tid];
            ds[tid + 256] = gs[tid + 256];
        }
#endif
        __syncthreads();
        const unsigned short* Bb = &ldsB[kt & 1][0];
        v16bf a0 = load_frag(&ldsA[(wm * 2 + 0) * 512], lane);
        v16bf a1 = load_frag(&ldsA[(wm * 2 + 1) * 512], lane);
#pragma unroll
        for (int j = 0; j < 4; ++j) {
            v16bf b = load_frag(&Bb[(wn * 4 + j) * 512], lane);
            acc[0][j] = __builtin_amdgcn_wmma_f32_16x16x32_bf16(
                false, a0, false, b, (short)0, acc[0][j], false, false);
            acc[1][j] = __builtin_amdgcn_wmma_f32_16x16x32_bf16(
                false, a1, false, b, (short)0, acc[1][j], false, false);
        }
    }

    // epilogue: +b1, exact GELU, store bf16 h at compact row
#pragma unroll
    for (int i = 0; i < 2; ++i) {
#pragma unroll
        for (int j = 0; j < 4; ++j) {
            int col = nbase + wn * 64 + j * 16 + (lane & 15);
            float bias = b1[e * DDIM + col];
            v8f c = acc[i][j];
#pragma unroll
            for (int r = 0; r < 8; ++r) {
                int lm = wm * 32 + i * 16 + r + 8 * (lane >> 4);
                int gm = mbase + lm;
                if (gm < cnt) {
                    float v = c[r] + bias;
                    v = 0.5f * v * (1.0f + erff(v * 0.70710678118f));
                    hbuf[(size_t)(off_e + gm) * DDIM + col] = f2bf(v);
                }
            }
        }
    }
}

// --------------------------------------- GEMM2: out += wt * (h W2 + b2)
__global__ __launch_bounds__(256)
void gemm2_kernel(const unsigned short* __restrict__ hbuf,
                  const unsigned short* __restrict__ w2s,
                  const float* __restrict__ b2,
                  float* __restrict__ out,
                  const int* __restrict__ rowtok,
                  const float* __restrict__ rowwt,
                  const int* __restrict__ counts,
                  const int* __restrict__ offs) {
    const int e = blockIdx.z;
    const int cnt = counts[e];
    const int mbase = blockIdx.x * TM;
    if (mbase >= cnt) return;
    const int nbase = blockIdx.y * TN;
    const int off_e = offs[e];

    __shared__ unsigned short ldsA[8 * 512];
    __shared__ unsigned short ldsB[2][8 * 512];
    __shared__ int   ridx[TM];
    __shared__ float rwt[TM];

    const int tid = threadIdx.x;
    const int lane = tid & 31, wave = tid >> 5;
    const int wm = wave & 3, wn = wave >> 2;

    if (tid < TM) {
        int gm = mbase + tid;
        int src = off_e + (gm < cnt ? gm : 0);
        ridx[tid] = rowtok[src];
        rwt[tid]  = rowwt[src];
    }

    const v8f vzero = {0.f, 0.f, 0.f, 0.f, 0.f, 0.f, 0.f, 0.f};
    v8f acc[2][4];
#pragma unroll
    for (int i = 0; i < 2; ++i)
#pragma unroll
        for (int j = 0; j < 4; ++j) acc[i][j] = vzero;

    const size_t btile = (size_t)e * KT2 * NT2 + (nbase >> 4);
#if HAVE_TDM
    if (wave == 0) tdm_load_8k(&ldsB[0][0], w2s + btile * 512);
#endif

    for (int kt = 0; kt < KT2; ++kt) {
        __syncthreads();
        {   // stage A: contiguous compact h rows, fragment order
            int r = tid >> 2;
            int kc = (tid & 3) << 3;
            int half = kc >> 4, khalf = (kc >> 3) & 1;
#pragma unroll
            for (int rep = 0; rep < 2; ++rep) {
                int rr = r + rep * 64;
                uint4 v = *(const uint4*)(hbuf + (size_t)(off_e + mbase + rr) * DDIM + kt * 32 + kc);
                int mt = rr >> 4, m = rr & 15;
                *(uint4*)(&ldsA[mt * 512 + half * 256 + (khalf * 16 + m) * 8]) = v;
            }
        }
#if HAVE_TDM
        if (wave == 0) {
            if (kt + 1 < KT2) {
                tdm_load_8k(&ldsB[(kt + 1) & 1][0], w2s + (btile + (size_t)(kt + 1) * NT2) * 512);
                __builtin_amdgcn_s_wait_tensorcnt(1);
            } else {
                __builtin_amdgcn_s_wait_tensorcnt(0);
            }
        }
#else
        {
            const uint4* gs = (const uint4*)(w2s + (btile + (size_t)kt * NT2) * 512);
            uint4* ds = (uint4*)(&ldsB[kt & 1][0]);
            ds[tid] = gs[tid];
            ds[tid + 256] = gs[tid + 256];
        }
#endif
        __syncthreads();
        const unsigned short* Bb = &ldsB[kt & 1][0];
        v16bf a0 = load_frag(&ldsA[(wm * 2 + 0) * 512], lane);
        v16bf a1 = load_frag(&ldsA[(wm * 2 + 1) * 512], lane);
#pragma unroll
        for (int j = 0; j < 4; ++j) {
            v16bf b = load_frag(&Bb[(wn * 4 + j) * 512], lane);
            acc[0][j] = __builtin_amdgcn_wmma_f32_16x16x32_bf16(
                false, a0, false, b, (short)0, acc[0][j], false, false);
            acc[1][j] = __builtin_amdgcn_wmma_f32_16x16x32_bf16(
                false, a1, false, b, (short)0, acc[1][j], false, false);
        }
    }

#pragma unroll
    for (int i = 0; i < 2; ++i) {
#pragma unroll
        for (int j = 0; j < 4; ++j) {
            int col = nbase + wn * 64 + j * 16 + (lane & 15);
            float bias = b2[e * CDIM + col];
            v8f c = acc[i][j];
#pragma unroll
            for (int r = 0; r < 8; ++r) {
                int lm = wm * 32 + i * 16 + r + 8 * (lane >> 4);
                int gm = mbase + lm;
                if (gm < cnt) {
                    int tok = ridx[lm];
                    float v = rwt[lm] * (c[r] + bias);
                    unsafeAtomicAdd(&out[(size_t)tok * CDIM + col], v);  // global_atomic_add_f32
                }
            }
        }
    }
}

// ------------------------------------------------------------------ launch
extern "C" void kernel_launch(void* const* d_in, const int* in_sizes, int n_in,
                              void* d_out, int out_size, void* d_ws, size_t ws_size,
                              hipStream_t stream) {
    const float* x   = (const float*)d_in[0];
    const float* rw  = (const float*)d_in[1];
    const float* W1  = (const float*)d_in[2];
    const float* b1  = (const float*)d_in[3];
    const float* W2  = (const float*)d_in[4];
    const float* b2  = (const float*)d_in[5];
    float* out = (float*)d_out;

    char* p = (char*)d_ws;
    auto alloc = [&](size_t bytes) -> void* {
        void* r = (void*)p;
        p += (bytes + 1023) & ~(size_t)1023;   // keep 1KB tile alignment
        return r;
    };
    unsigned short* xbf  = (unsigned short*)alloc((size_t)NTOK * CDIM * 2);
    unsigned short* w1s  = (unsigned short*)alloc((size_t)NEXP * CDIM * DDIM * 2);
    unsigned short* w2s  = (unsigned short*)alloc((size_t)NEXP * DDIM * CDIM * 2);
    unsigned short* hbuf = (unsigned short*)alloc((size_t)(ROWS + TM) * DDIM * 2);
    int*   rowtok = (int*)alloc((size_t)ROWS * 4);
    float* rowwt  = (float*)alloc((size_t)ROWS * 4);
    int*   t2e    = (int*)alloc((size_t)NTOK * 2 * 4);
    float* t2w    = (float*)alloc((size_t)NTOK * 2 * 4);
    int*   counts = (int*)alloc(1024);
    int*   offs   = (int*)alloc(1024);
    int*   fill   = (int*)alloc(1024);

    zero_kernel<<<2048, 256, 0, stream>>>(out, (size_t)out_size, counts);
    convert_kernel<<<4096, 256, 0, stream>>>(x, W1, W2, xbf, w1s, w2s);
    router_kernel<<<NTOK / 8, 256, 0, stream>>>(x, rw, t2e, t2w, counts);
    scan_kernel<<<1, 32, 0, stream>>>(counts, offs, fill);
    assign_kernel<<<NTOK / 256, 256, 0, stream>>>(t2e, t2w, offs, fill, rowtok, rowwt);
    gemm1_kernel<<<dim3(NTOK / TM, DDIM / TN, NEXP), 256, 0, stream>>>(
        xbf, w1s, b1, hbuf, rowtok, counts, offs);
    gemm2_kernel<<<dim3(NTOK / TM, CDIM / TN, NEXP), 256, 0, stream>>>(
        hbuf, w2s, b2, out, rowtok, rowwt, counts, offs);
}
